// Relation_Latent_Encoder_68719476884
// MI455X (gfx1250) — matrix operand order repaired and checked by reading, not compiled, not run on hardware
//
#include <hip/hip_runtime.h>
#include <hip/hip_bf16.h>
#include <math.h>

// ---------------------------------------------------------------------------
// Problem constants (from setup_inputs; deterministic)
// ---------------------------------------------------------------------------
#define B_    32
#define N_    512
#define C_    1024
#define H_    16
#define D_    64
#define MTOT  (B_ * N_)     // 16384 rows
#define QKVN  (3 * C_)      // 3072
#define LAT   64

typedef __attribute__((ext_vector_type(16))) __bf16 v16bf;
typedef __attribute__((ext_vector_type(8)))  __bf16 v8bf;
typedef __attribute__((ext_vector_type(4)))  __bf16 v4bf;
typedef __attribute__((ext_vector_type(8)))  float  v8f;

// ---------------------------------------------------------------------------
// WMMA helpers: D = A(16x32 bf16) x B(32x16 bf16) + C(16x16 f32)
// ---------------------------------------------------------------------------
__device__ __forceinline__ v8f wmma_bf16(v16bf a, v16bf b, v8f c) {
  return __builtin_amdgcn_wmma_f32_16x16x32_bf16(
      /*neg_a=*/false, a, /*neg_b=*/false, b,
      /*c_mod=*/(short)0, c, /*reuse_a=*/false, /*reuse_b=*/false);
}

// A-matrix fragment (16x32, row-major source, ld in elements).
// Layout used here (contiguous k): lane<16 -> row (lane), K = k0+0..15
//             lane>=16 -> row (lane-16), K = k0+16..31
__device__ __forceinline__ v16bf load_a_frag(const __bf16* base, int ld,
                                             int row0, int k0, int lane) {
  int r  = row0 + (lane & 15);
  int kk = k0 + ((lane & 16) ? 16 : 0);
  const __bf16* p = base + (size_t)r * ld + kk;
  v8bf lo = *reinterpret_cast<const v8bf*>(p);
  v8bf hi = *reinterpret_cast<const v8bf*>(p + 8);
  return __builtin_shufflevector(lo, hi, 0, 1, 2, 3, 4, 5, 6, 7,
                                 8, 9, 10, 11, 12, 13, 14, 15);
}

// B-matrix fragment (32x16) from "column-major-friendly" storage: baseT is
// row-major [Ncols][K] so each output column's K values are contiguous.
// Layout used here (contiguous k): lane<16 -> col (lane), K = k0+0..15 ; lane>=16 -> K = k0+16..31
__device__ __forceinline__ v16bf load_b_frag(const __bf16* baseT, int ld,
                                             int col0, int k0, int lane) {
  int c  = col0 + (lane & 15);
  int kk = k0 + ((lane & 16) ? 16 : 0);
  return *reinterpret_cast<const v16bf*>(baseT + (size_t)c * ld + kk);
}

// ---------------------------------------------------------------------------
// 1) Cast + transpose qkv_w (CxQKVN fp32) -> WT (QKVN x C bf16)
// ---------------------------------------------------------------------------
__global__ __launch_bounds__(256) void cast_tr_w_kernel(
    const float* __restrict__ W, __bf16* __restrict__ WT) {
  size_t idx = (size_t)blockIdx.x * 256 + threadIdx.x;  // over C_*QKVN
  int k = (int)(idx / QKVN);
  int n = (int)(idx % QKVN);
  WT[(size_t)n * C_ + k] = (__bf16)W[idx];
}

// ---------------------------------------------------------------------------
// 2) LayerNorm each row (1024 wide) -> bf16
// ---------------------------------------------------------------------------
__global__ __launch_bounds__(256) void ln_kernel(
    const float* __restrict__ r, const float* __restrict__ g,
    const float* __restrict__ be, __bf16* __restrict__ x) {
  const int row  = blockIdx.x;
  const int t    = threadIdx.x;
  const int lane = t & 31;
  const int wid  = t >> 5;

  float4 v4 = *reinterpret_cast<const float4*>(r + (size_t)row * C_ + t * 4);
  float vv[4] = {v4.x, v4.y, v4.z, v4.w};
  float s = 0.f, ss = 0.f;
#pragma unroll
  for (int i = 0; i < 4; ++i) { s += vv[i]; ss += vv[i] * vv[i]; }
#pragma unroll
  for (int m = 1; m < 32; m <<= 1) {
    s  += __shfl_xor(s, m, 32);
    ss += __shfl_xor(ss, m, 32);
  }
  __shared__ float red[8][2];
  if (lane == 0) { red[wid][0] = s; red[wid][1] = ss; }
  __syncthreads();
  __shared__ float stat[2];
  if (t == 0) {
    float S = 0.f, SS = 0.f;
#pragma unroll
    for (int i = 0; i < 8; ++i) { S += red[i][0]; SS += red[i][1]; }
    float mean = S * (1.0f / C_);
    float var  = SS * (1.0f / C_) - mean * mean;
    stat[0] = mean;
    stat[1] = rsqrtf(var + 1e-5f);
  }
  __syncthreads();
  float mean = stat[0], rstd = stat[1];
  v4bf ov;
#pragma unroll
  for (int i = 0; i < 4; ++i) {
    int c = t * 4 + i;
    ov[i] = (__bf16)((vv[i] - mean) * rstd * g[c] + be[c]);
  }
  *reinterpret_cast<v4bf*>(x + (size_t)row * C_ + t * 4) = ov;
}

// ---------------------------------------------------------------------------
// 3) QKV GEMM: qkv = X(16384x1024 bf16) @ W(1024x3072) + b, stored bf16.
//    Block = 256 thr (8 waves as 4M x 2N). Block tile 128x128,
//    wave tile 32x64 -> 8 wmma accumulators. K step 32.
// ---------------------------------------------------------------------------
__global__ __launch_bounds__(256) void gemm_qkv_kernel(
    const __bf16* __restrict__ X, const __bf16* __restrict__ WT,
    const float* __restrict__ bias, __bf16* __restrict__ out) {
  const int lane = threadIdx.x & 31;
  const int wid  = threadIdx.x >> 5;
  const int wm = wid & 3;   // 0..3
  const int wn = wid >> 2;  // 0..1
  const int row0 = blockIdx.x * 128 + wm * 32;
  const int col0 = blockIdx.y * 128 + wn * 64;

  v8f acc[2][4] = {};
  for (int k0 = 0; k0 < C_; k0 += 32) {
    v16bf a0 = load_a_frag(X, C_, row0, k0, lane);
    v16bf a1 = load_a_frag(X, C_, row0 + 16, k0, lane);
    v16bf bf[4];
#pragma unroll
    for (int ni = 0; ni < 4; ++ni)
      bf[ni] = load_b_frag(WT, C_, col0 + ni * 16, k0, lane);
#pragma unroll
    for (int ni = 0; ni < 4; ++ni) {
      acc[0][ni] = wmma_bf16(a0, bf[ni], acc[0][ni]);
      acc[1][ni] = wmma_bf16(a1, bf[ni], acc[1][ni]);
    }
  }
  // epilogue: +bias, store bf16 (C layout: lane<16 rows j, lane>=16 rows j+8)
#pragma unroll
  for (int mi = 0; mi < 2; ++mi) {
#pragma unroll
    for (int ni = 0; ni < 4; ++ni) {
      int col = col0 + ni * 16 + (lane & 15);
      float bia = bias[col];
#pragma unroll
      for (int j = 0; j < 8; ++j) {
        int rowin = (lane & 16) ? (j + 8) : j;
        size_t row = (size_t)row0 + mi * 16 + rowin;
        out[row * QKVN + col] = (__bf16)(acc[mi][ni][j] + bia);
      }
    }
  }
}

// ---------------------------------------------------------------------------
// 4) Flash attention. Block = 128 thr (4 waves), one (b,h,64-row tile).
//    Wave handles 16 query rows; online softmax; scale = D^-0.5 = 0.125.
// ---------------------------------------------------------------------------
__global__ __launch_bounds__(128) void attn_kernel(
    const __bf16* __restrict__ qkv, __bf16* __restrict__ attnout) {
  __shared__ __bf16 VT[D_][64];       // [d][key] transposed V tile
  __shared__ __bf16 PB[4][16][64];    // per-wave P staging (A-layout source)

  const int tid  = threadIdx.x;
  const int lane = tid & 31;
  const int w    = tid >> 5;
  const int blk  = blockIdx.x;        // b*128 + h*8 + mt
  const int mt   = blk & 7;
  const int h    = (blk >> 3) & (H_ - 1);
  const int b    = blk >> 7;

  const int rowQ0 = b * N_ + mt * 64 + w * 16;
  const int qcol  = h * D_;

  // Q fragments (A-layout), K dim = d (64 -> two 32-steps)
  v16bf qf0 = load_a_frag(qkv + qcol, QKVN, rowQ0, 0, lane);
  v16bf qf1 = load_a_frag(qkv + qcol, QKVN, rowQ0, 32, lane);

  v8f o[4] = {};
  float m[8], l[8];
#pragma unroll
  for (int j = 0; j < 8; ++j) { m[j] = -1e30f; l[j] = 0.f; }

  const __bf16* kbase = qkv + C_ + qcol;       // K section
  for (int jt = 0; jt < N_ / 64; ++jt) {
    __syncthreads();
    // cooperative transposed V tile load: V[key][d] -> VT[d][key]
    for (int i = tid; i < 64 * 64; i += 128) {
      int key = i >> 6, d = i & 63;
      VT[d][key] =
          qkv[(size_t)(b * N_ + jt * 64 + key) * QKVN + 2 * C_ + h * D_ + d];
    }
    __syncthreads();

    // S = Q K^T * scale  (4 subtiles of 16 keys)
    v8f s[4];
#pragma unroll
    for (int t = 0; t < 4; ++t) {
      int keyrow0 = b * N_ + jt * 64 + t * 16;
      v16bf kf0 = load_b_frag(kbase, QKVN, keyrow0, 0, lane);
      v16bf kf1 = load_b_frag(kbase, QKVN, keyrow0, 32, lane);
      v8f z = {};
      z = wmma_bf16(qf0, kf0, z);
      z = wmma_bf16(qf1, kf1, z);
#pragma unroll
      for (int j = 0; j < 8; ++j) z[j] *= 0.125f;
      s[t] = z;
    }

    // online softmax: row max across 4 subtiles + 16-lane xor reduce
    float mn[8], alpha[8], rs[8];
#pragma unroll
    for (int j = 0; j < 8; ++j) {
      float v = fmaxf(fmaxf(s[0][j], s[1][j]), fmaxf(s[2][j], s[3][j]));
      v = fmaxf(v, __shfl_xor(v, 1, 32));
      v = fmaxf(v, __shfl_xor(v, 2, 32));
      v = fmaxf(v, __shfl_xor(v, 4, 32));
      v = fmaxf(v, __shfl_xor(v, 8, 32));
      mn[j]    = fmaxf(m[j], v);
      alpha[j] = __expf(m[j] - mn[j]);
      m[j]     = mn[j];
      rs[j]    = 0.f;
    }
    // P = exp(S - mnew), stage to LDS, accumulate row sums
#pragma unroll
    for (int t = 0; t < 4; ++t) {
      int col = t * 16 + (lane & 15);
#pragma unroll
      for (int j = 0; j < 8; ++j) {
        float p = __expf(s[t][j] - mn[j]);
        rs[j] += p;
        int rowin = (lane & 16) ? (j + 8) : j;
        PB[w][rowin][col] = (__bf16)p;
      }
    }
#pragma unroll
    for (int j = 0; j < 8; ++j) {
      float v = rs[j];
      v += __shfl_xor(v, 1, 32);
      v += __shfl_xor(v, 2, 32);
      v += __shfl_xor(v, 4, 32);
      v += __shfl_xor(v, 8, 32);
      l[j] = l[j] * alpha[j] + v;
    }
#pragma unroll
    for (int t = 0; t < 4; ++t)
#pragma unroll
      for (int j = 0; j < 8; ++j) o[t][j] *= alpha[j];

    // O += P @ V  (KK = keys, two 32-steps; 4 d-subtiles)
    v16bf pa0 = load_a_frag(&PB[w][0][0], 64, 0, 0, lane);
    v16bf pa1 = load_a_frag(&PB[w][0][0], 64, 0, 32, lane);
#pragma unroll
    for (int t = 0; t < 4; ++t) {
      v16bf vf0 = load_b_frag(&VT[0][0], 64, t * 16, 0, lane);
      v16bf vf1 = load_b_frag(&VT[0][0], 64, t * 16, 32, lane);
      o[t] = wmma_bf16(pa0, vf0, o[t]);
      o[t] = wmma_bf16(pa1, vf1, o[t]);
    }
  }
  // finalize: divide by row sum, store bf16
#pragma unroll
  for (int t = 0; t < 4; ++t) {
#pragma unroll
    for (int j = 0; j < 8; ++j) {
      int rowin = (lane & 16) ? (j + 8) : j;
      size_t row = (size_t)rowQ0 + rowin;
      int col = h * D_ + t * 16 + (lane & 15);
      attnout[row * C_ + col] = (__bf16)(o[t][j] / l[j]);
    }
  }
}

// ---------------------------------------------------------------------------
// 5) Column sum of attn output (proj GEMM folded into a GEMV on the mean row)
// ---------------------------------------------------------------------------
__global__ __launch_bounds__(256) void zero_kernel(float* p) {
  p[blockIdx.x * 256 + threadIdx.x] = 0.f;
}

__global__ __launch_bounds__(256) void colsum_kernel(
    const __bf16* __restrict__ a, float* __restrict__ colsum) {
  int c  = blockIdx.x * 256 + threadIdx.x;  // grid.x = 4
  int r0 = blockIdx.y * 256;                // grid.y = 64
  float s = 0.f;
  for (int i = 0; i < 256; ++i) s += (float)a[(size_t)(r0 + i) * C_ + c];
  atomicAdd(&colsum[c], s);
}

// mu = mean_{B,N}(2 * (attn @ proj_w + proj_b)) = 2*((colsum/M) @ W + b)
__global__ __launch_bounds__(256) void mu_kernel(
    const float* __restrict__ colsum, const float* __restrict__ pw,
    const float* __restrict__ pb, float* __restrict__ mu) {
  int c = blockIdx.x * 256 + threadIdx.x;
  float acc = 0.f;
  for (int k = 0; k < C_; ++k) acc += colsum[k] * pw[(size_t)k * C_ + c];
  mu[c] = 2.f * (acc * (1.0f / MTOT) + pb[c]);
}

__global__ __launch_bounds__(256) void sigma_kernel(
    const float* __restrict__ mu, const float* __restrict__ sw,
    const float* __restrict__ sb, float* __restrict__ sigma) {
  int c = blockIdx.x * 256 + threadIdx.x;
  float acc = 0.f;
  for (int k = 0; k < C_; ++k) acc += mu[k] * sw[(size_t)k * C_ + c];
  float t = acc + sb[c];
  sigma[c] = 0.1f + 0.9f / (1.f + __expf(-t));
}

// ---------------------------------------------------------------------------
// 6) latent = mu + sigma * eps, eps = threefry2x32(key 42) -> erfinv normal
// ---------------------------------------------------------------------------
__device__ __forceinline__ float erfinv_f(float u) {
  float w = -__logf((1.0f - u) * (1.0f + u));
  float p;
  if (w < 5.0f) {
    w -= 2.5f;
    p = 2.81022636e-08f;
    p = fmaf(p, w, 3.43273939e-07f);
    p = fmaf(p, w, -3.5233877e-06f);
    p = fmaf(p, w, -4.39150654e-06f);
    p = fmaf(p, w, 0.00021858087f);
    p = fmaf(p, w, -0.00125372503f);
    p = fmaf(p, w, -0.00417768164f);
    p = fmaf(p, w, 0.246640727f);
    p = fmaf(p, w, 1.50140941f);
  } else {
    w = sqrtf(w) - 3.0f;
    p = -0.000200214257f;
    p = fmaf(p, w, 0.000100950558f);
    p = fmaf(p, w, 0.00134934322f);
    p = fmaf(p, w, -0.00367342844f);
    p = fmaf(p, w, 0.00573950773f);
    p = fmaf(p, w, -0.0076224613f);
    p = fmaf(p, w, 0.00943887047f);
    p = fmaf(p, w, 1.00167406f);
    p = fmaf(p, w, 2.83297682f);
  }
  return p * u;
}

__global__ __launch_bounds__(256) void latent_kernel(
    const float* __restrict__ mu, const float* __restrict__ sigma,
    float* __restrict__ out) {
  const int idx = blockIdx.x * 256 + threadIdx.x;  // 65536 total
  const int c = idx & (C_ - 1);
  // threefry2x32, key = (0, 42), counters split [0,32768) / [32768,65536)
  const unsigned j = (unsigned)idx & 32767u;
  const bool hihalf = idx >= 32768;
  const unsigned k0 = 0u, k1 = 42u, k2 = 0x1BD11BDAu ^ k0 ^ k1;
  unsigned x0 = j + k0;
  unsigned x1 = (j + 32768u) + k1;
#define TFR(r)                            \
  {                                       \
    x0 += x1;                             \
    x1 = (x1 << (r)) | (x1 >> (32 - (r))); \
    x1 ^= x0;                             \
  }
  TFR(13) TFR(15) TFR(26) TFR(6)  x0 += k1; x1 += k2 + 1u;
  TFR(17) TFR(29) TFR(16) TFR(24) x0 += k2; x1 += k0 + 2u;
  TFR(13) TFR(15) TFR(26) TFR(6)  x0 += k0; x1 += k1 + 3u;
  TFR(17) TFR(29) TFR(16) TFR(24) x0 += k1; x1 += k2 + 4u;
  TFR(13) TFR(15) TFR(26) TFR(6)  x0 += k2; x1 += k0 + 5u;
#undef TFR
  unsigned bits = hihalf ? x1 : x0;
  float f = __uint_as_float(0x3f800000u | (bits >> 9)) - 1.0f;  // [0,1)
  const float lo = -0.99999994f;
  float u = fmaxf(f * (1.0f - lo) + lo, lo);                    // (-1,1)
  float eps = 1.41421356f * erfinv_f(u);
  out[idx] = mu[c] + sigma[c] * eps;
}

// ---------------------------------------------------------------------------
// Launch
// ---------------------------------------------------------------------------
extern "C" void kernel_launch(void* const* d_in, const int* in_sizes, int n_in,
                              void* d_out, int out_size, void* d_ws,
                              size_t ws_size, hipStream_t stream) {
  const float* r     = (const float*)d_in[0];
  const float* lng   = (const float*)d_in[1];
  const float* lnb   = (const float*)d_in[2];
  const float* qkvw  = (const float*)d_in[3];
  const float* qkvb  = (const float*)d_in[4];
  const float* projw = (const float*)d_in[5];
  const float* projb = (const float*)d_in[6];
  const float* sigw  = (const float*)d_in[7];
  const float* sigb  = (const float*)d_in[8];
  float* out = (float*)d_out;

  char* ws = (char*)d_ws;
  size_t off = 0;
  __bf16* x    = (__bf16*)(ws + off); off += (size_t)MTOT * C_ * 2;   // 32 MB
  __bf16* wT   = (__bf16*)(ws + off); off += (size_t)QKVN * C_ * 2;   // 6 MB
  __bf16* qkv  = (__bf16*)(ws + off); off += (size_t)MTOT * QKVN * 2; // 96 MB
  __bf16* aout = (__bf16*)(ws + off); off += (size_t)MTOT * C_ * 2;   // 32 MB
  float* colsum = (float*)(ws + off); off += C_ * sizeof(float);
  float* mu     = (float*)(ws + off); off += C_ * sizeof(float);
  float* sigma  = (float*)(ws + off); off += C_ * sizeof(float);

  cast_tr_w_kernel<<<(C_ * QKVN) / 256, 256, 0, stream>>>(qkvw, wT);
  ln_kernel<<<MTOT, 256, 0, stream>>>(r, lng, lnb, x);
  gemm_qkv_kernel<<<dim3(MTOT / 128, QKVN / 128), 256, 0, stream>>>(
      x, wT, qkvb, qkv);
  attn_kernel<<<B_ * H_ * (N_ / 64), 128, 0, stream>>>(qkv, aout);
  zero_kernel<<<C_ / 256, 256, 0, stream>>>(colsum);
  colsum_kernel<<<dim3(C_ / 256, MTOT / 256), 256, 0, stream>>>(aout, colsum);
  mu_kernel<<<C_ / 256, 256, 0, stream>>>(colsum, projw, projb, mu);
  sigma_kernel<<<C_ / 256, 256, 0, stream>>>(mu, sigw, sigb, sigma);
  latent_kernel<<<(LAT * C_) / 256, 256, 0, stream>>>(mu, sigma, out);
}
